// EarthSpecificLayer_83202106458383
// MI455X (gfx1250) — matrix operand order repaired and checked by reading, not compiled, not run on hardware
//
#include <hip/hip_runtime.h>
#include <hip/hip_bf16.h>

// ---------------------------------------------------------------------------
// 3D Swin block (Pangu EarthSpecificLayer), DEPTH=2, all GEMMs via
// v_wmma_f32_16x16x32_bf16 (bf16 in, f32 accumulate).
// Weights are pre-packed once per layer into bf16 WMMA-B layout ([n][k],
// K-contiguous) so every B-tile fetch is two global_load_b128 per lane.
// Shapes: B=1 Z=8 H=48 W=192 C=192 heads=6 hd=32 win=(2,6,12) N=144
// windows=512 (Twin=32 x nw=16), tokens=73728.
// ---------------------------------------------------------------------------

typedef __attribute__((ext_vector_type(16))) __bf16 v16bf;
typedef __attribute__((ext_vector_type(8)))  float  v8f;
typedef unsigned short u16;
typedef unsigned int   u32;

#define ZD 8
#define HD_ 48
#define WD 192
#define CD 192
#define NHEAD 6
#define HDIM 32
#define NWIN 144        // tokens per window
#define NWP 160         // padded token dim (K for P@V)
#define TWIN 32
#define WINTOT 512
#define NTOK 73728
#define QSCALE 0.17677669529663689f  // 32^-0.5

union V16U { v16bf v; u16 s[16]; u32 u[8]; uint4 q[2]; };

__device__ __forceinline__ u16 f2bf(float f) {
  u32 u = __float_as_uint(f);
  u32 r = u + 0x7FFFu + ((u >> 16) & 1u);       // RNE
  if ((u & 0x7F800000u) == 0x7F800000u) r = u;  // inf/NaN passthrough
  return (u16)(r >> 16);
}

// A operand (16x32 bf16 tile) from row-major [row][k] bf16 buffer.
// lane L: M = row0 + (L&15); K base = (L<16 ? 0 : 16), 16 contiguous K.
__device__ __forceinline__ v16bf load_a_rm(const u16* base, int row0, int k0,
                                           int ldk, int lane) {
  const u16* r = base + (size_t)(row0 + (lane & 15)) * ldk;
  int kb = k0 + ((lane >> 4) << 4);
  V16U t;
#pragma unroll
  for (int rr = 0; rr < 8; ++rr) {
    int k = kb + 2 * rr;
    t.u[rr] = *(const u32*)(r + k);
  }
  return t.v;
}

// B operand from N-major bf16 buffer [n][k] (16 contiguous K per lane).
// Requires 32B alignment of (base + n*ldk + k) -- true for ldk in {32,160,192,768}
// with k0 a multiple of 16 and 32B-aligned base.
__device__ __forceinline__ v16bf load_b_nk(const u16* base, int n0, int k0,
                                           int ldk, int lane) {
  const u16* p = base + (size_t)(n0 + (lane & 15)) * ldk + k0 + ((lane >> 4) << 4);
  V16U t;
  t.q[0] = *(const uint4*)p;
  t.q[1] = *(const uint4*)(p + 8);
  return t.v;
}

__device__ __forceinline__ v8f wmma_bf16(v16bf a, v16bf b, v8f c) {
  return __builtin_amdgcn_wmma_f32_16x16x32_bf16(false, a, false, b, (short)0, c,
                                                 false, false);
}

// ---------------------------------------------------------------------------
__global__ void k_copy(float* dst, const float* __restrict__ src, int n) {
  for (int i = blockIdx.x * blockDim.x + threadIdx.x; i < n;
       i += gridDim.x * blockDim.x)
    dst[i] = src[i];
}

// Pack fp32 weight [K][N] (row-major) -> bf16 [N][K] (WMMA-B layout).
__global__ void k_packw(const float* __restrict__ W, u16* out, int K, int N) {
  int total = K * N;
  for (int idx = blockIdx.x * blockDim.x + threadIdx.x; idx < total;
       idx += gridDim.x * blockDim.x) {
    int n = idx / K, k = idx % K;
    out[idx] = f2bf(W[(size_t)k * N + n]);
  }
}

// ---------------------------------------------------------------------------
// QKV: per-window gather (+roll) -> LDS bf16, GEMM (144x576, K=192),
// split epilogue into q*scale / k / v^T(padded).
template <bool ROLLED>
__global__ void k_qkv(const float* __restrict__ x, const u16* __restrict__ Wq,
                      const float* __restrict__ Bq, u16* qs, u16* ks, u16* vtp) {
  extern __shared__ char smem[];
  u16* Xw = (u16*)smem;  // 144 x 192 bf16
  const int win = blockIdx.x;
  const int t = win >> 4, tw = win & 15;
  const int tz = t >> 3, th = t & 7;
  const int tid = threadIdx.x, lane = tid & 31, wid = tid >> 5;
  const int nwv = blockDim.x >> 5;

  // stage window tokens
  for (int idx = tid; idx < NWIN * CD; idx += blockDim.x) {
    int n = idx / CD, c = idx % CD;
    int lz = n / 72, lh = (n / 12) % 6, lw = n % 12;
    int z = tz * 2 + lz, h = th * 6 + lh, w = tw * 12 + lw;
    if (ROLLED) { z = (z + 7) & 7; h = (h + 45) % 48; w = (w + 186) % 192; }
    Xw[idx] = f2bf(x[((size_t)(z * HD_ + h) * WD + w) * CD + c]);
  }
  // zero v^T pad columns (tokens 144..159)
  for (int idx = tid; idx < NHEAD * HDIM * 16; idx += blockDim.x) {
    int head = idx >> 9, rem = idx & 511, d = rem >> 4, n = 144 + (rem & 15);
    vtp[((size_t)(win * NHEAD + head) * HDIM + d) * NWP + n] = 0;
  }
  __syncthreads();

  // 9 x 36 output tiles, K = 192 (6 wmma)
  for (int tile = wid; tile < 9 * 36; tile += nwv) {
    int tr = tile / 36, tc = tile % 36;
    v8f acc = {};
#pragma unroll
    for (int kk = 0; kk < 6; ++kk) {
      v16bf a = load_a_rm(Xw, tr * 16, kk * 32, CD, lane);
      v16bf b = load_b_nk(Wq, tc * 16, kk * 32, CD, lane);
      acc = wmma_bf16(a, b, acc);
    }
#pragma unroll
    for (int r = 0; r < 8; ++r) {
      int n = tr * 16 + r + ((lane >> 4) << 3);
      int c = tc * 16 + (lane & 15);
      float v = acc[r] + Bq[c];
      if (c < CD) {
        int head = c >> 5, d = c & 31;
        qs[((size_t)(win * NHEAD + head) * NWIN + n) * HDIM + d] = f2bf(v * QSCALE);
      } else if (c < 2 * CD) {
        int cc = c - CD, head = cc >> 5, d = cc & 31;
        ks[((size_t)(win * NHEAD + head) * NWIN + n) * HDIM + d] = f2bf(v);
      } else {
        int cc = c - 2 * CD, head = cc >> 5, d = cc & 31;
        vtp[((size_t)(win * NHEAD + head) * HDIM + d) * NWP + n] = f2bf(v);
      }
    }
  }
}

// ---------------------------------------------------------------------------
// Dense relative-position bias [Twin][heads][144][144] (+ shift mask if rolled)
__global__ void k_bias(const float* __restrict__ bt, float* biasM, int rolled) {
  const int thx = blockIdx.x;            // t*6 + head, 192 blocks
  const int t = thx / 6, hh = thx % 6;
  const int tz = t >> 3, th = t & 7;
  for (int idx = threadIdx.x; idx < NWIN * NWIN; idx += blockDim.x) {
    int n = idx / NWIN, m = idx % NWIN;
    int a = n / 72, b = (n / 12) % 6, c = n % 12;
    int a2 = m / 72, b2 = (m / 12) % 6, c2 = m % 12;
    int pos = (a + 2 * a2) * 828 + (b + 6 * b2) * 23 + (c - c2 + 11);
    float v = bt[((size_t)pos * TWIN + t) * NHEAD + hh];
    if (rolled) {
      int idn = ((tz * 2 + a) >= 1) * 2 + ((th * 6 + b) >= 3);
      int idm = ((tz * 2 + a2) >= 1) * 2 + ((th * 6 + b2) >= 3);
      if (idn != idm) v -= 10000.0f;
    }
    biasM[((size_t)thx * NWIN + n) * NWIN + m] = v;
  }
}

// ---------------------------------------------------------------------------
// Attention: one block per (window, head); 9 waves, each owns a 16-row strip.
__global__ void k_attn(const u16* __restrict__ qs, const u16* __restrict__ ks,
                       const u16* __restrict__ vtp, const float* __restrict__ biasM,
                       u16* ao) {
  extern __shared__ char smem[];
  const int blk = blockIdx.x;
  const int win = blk / NHEAD, head = blk % NHEAD;
  const int t = win >> 4;
  const int tid = threadIdx.x, lane = tid & 31, wid = tid >> 5;  // wid 0..8
  u16* P = (u16*)smem + (size_t)wid * 16 * NWP;

  const u16* q = qs + (size_t)(win * NHEAD + head) * NWIN * HDIM;
  const u16* k = ks + (size_t)(win * NHEAD + head) * NWIN * HDIM;
  const u16* vt = vtp + (size_t)(win * NHEAD + head) * HDIM * NWP;
  const float* bm = biasM + (size_t)(t * NHEAD + head) * NWIN * NWIN;
  const int n0 = wid * 16;

  // S strip = q[n0:n0+16] @ k^T   (hd=32 == one K step)
  v8f s[9];
  {
    v16bf a = load_a_rm(q, n0, 0, HDIM, lane);
#pragma unroll
    for (int mt = 0; mt < 9; ++mt) {
      v16bf b = load_b_nk(k, mt * 16, 0, HDIM, lane);
      v8f z = {};
      s[mt] = wmma_bf16(a, b, z);
    }
  }
  // + bias
#pragma unroll
  for (int mt = 0; mt < 9; ++mt)
#pragma unroll
    for (int r = 0; r < 8; ++r) {
      int row = n0 + r + ((lane >> 4) << 3);
      s[mt][r] += bm[(size_t)row * NWIN + mt * 16 + (lane & 15)];
    }
  // row softmax (rows live across 16 lanes in same half-wave)
#pragma unroll
  for (int r = 0; r < 8; ++r) {
    float mx = -1e30f;
#pragma unroll
    for (int mt = 0; mt < 9; ++mt) mx = fmaxf(mx, s[mt][r]);
    for (int o = 1; o < 16; o <<= 1) mx = fmaxf(mx, __shfl_xor(mx, o, 32));
    float sum = 0.0f;
#pragma unroll
    for (int mt = 0; mt < 9; ++mt) {
      float e = __expf(s[mt][r] - mx);
      s[mt][r] = e;
      sum += e;
    }
    for (int o = 1; o < 16; o <<= 1) sum += __shfl_xor(sum, o, 32);
    float inv = 1.0f / sum;
#pragma unroll
    for (int mt = 0; mt < 9; ++mt) s[mt][r] *= inv;
  }
  // stage P (C layout -> row-major, zero-padded K to 160)
  for (int i = lane; i < 16 * 16; i += 32) P[(i >> 4) * NWP + 144 + (i & 15)] = 0;
#pragma unroll
  for (int mt = 0; mt < 9; ++mt)
#pragma unroll
    for (int r = 0; r < 8; ++r)
      P[(r + ((lane >> 4) << 3)) * NWP + mt * 16 + (lane & 15)] = f2bf(s[mt][r]);
  __syncthreads();

  // O strip = P @ v   (2 column tiles of 16, K = 160)
  v8f o0 = {}, o1 = {};
#pragma unroll
  for (int kk = 0; kk < 5; ++kk) {
    v16bf a = load_a_rm(P, 0, kk * 32, NWP, lane);
    o0 = wmma_bf16(a, load_b_nk(vt, 0, kk * 32, NWP, lane), o0);
    o1 = wmma_bf16(a, load_b_nk(vt, 16, kk * 32, NWP, lane), o1);
  }
#pragma unroll
  for (int r = 0; r < 8; ++r) {
    int row = n0 + r + ((lane >> 4) << 3);
    int c0 = head * HDIM + (lane & 15);
    ao[((size_t)win * NWIN + row) * CD + c0] = f2bf(o0[r]);
    ao[((size_t)win * NWIN + row) * CD + c0 + 16] = f2bf(o1[r]);
  }
}

// ---------------------------------------------------------------------------
// Projection + window-reverse + LayerNorm + residual (in-place on x).
template <bool ROLLED>
__global__ void k_proj(const u16* __restrict__ ao, const u16* __restrict__ Wp,
                       const float* __restrict__ Bp, const float* __restrict__ g,
                       const float* __restrict__ bb, float* x) {
  extern __shared__ char smem[];
  float* Y = (float*)smem;          // 144 x 192
  float* mu = Y + NWIN * CD;
  float* rs = mu + NWIN;
  const int win = blockIdx.x;
  const int t = win >> 4, tw = win & 15;
  const int tz = t >> 3, th = t & 7;
  const int tid = threadIdx.x, lane = tid & 31, wid = tid >> 5;
  const int nwv = blockDim.x >> 5;
  const u16* A = ao + (size_t)win * NWIN * CD;

  for (int tile = wid; tile < 9 * 12; tile += nwv) {
    int tr = tile / 12, tc = tile % 12;
    v8f acc = {};
#pragma unroll
    for (int kk = 0; kk < 6; ++kk) {
      v16bf a = load_a_rm(A, tr * 16, kk * 32, CD, lane);
      v16bf b = load_b_nk(Wp, tc * 16, kk * 32, CD, lane);
      acc = wmma_bf16(a, b, acc);
    }
#pragma unroll
    for (int r = 0; r < 8; ++r) {
      int n = tr * 16 + r + ((lane >> 4) << 3);
      int c = tc * 16 + (lane & 15);
      Y[n * CD + c] = acc[r] + Bp[c];
    }
  }
  __syncthreads();
  if (tid < NWIN) {
    float m = 0.0f;
    for (int c = 0; c < CD; ++c) m += Y[tid * CD + c];
    m *= (1.0f / CD);
    float v = 0.0f;
    for (int c = 0; c < CD; ++c) {
      float d = Y[tid * CD + c] - m;
      v += d * d;
    }
    mu[tid] = m;
    rs[tid] = rsqrtf(v * (1.0f / CD) + 1e-5f);
  }
  __syncthreads();
  for (int idx = tid; idx < NWIN * CD; idx += blockDim.x) {
    int n = idx / CD, c = idx % CD;
    int lz = n / 72, lh = (n / 12) % 6, lw = n % 12;
    int z = tz * 2 + lz, h = th * 6 + lh, w = tw * 12 + lw;
    if (ROLLED) { z = (z + 7) & 7; h = (h + 45) % 48; w = (w + 186) % 192; }
    size_t tk = (size_t)(z * HD_ + h) * WD + w;
    float val = (Y[idx] - mu[n]) * rs[n] * g[c] + bb[c];
    x[tk * CD + c] = x[tk * CD + c] + val;
  }
}

// ---------------------------------------------------------------------------
// Fused MLP per 64-token strip: GELU(x@W1+b1)@W2+b2 -> LN -> residual.
__global__ void k_mlp(const float* __restrict__ xin, const u16* __restrict__ W1,
                      const float* __restrict__ B1, const u16* __restrict__ W2,
                      const float* __restrict__ B2, const float* __restrict__ g,
                      const float* __restrict__ bb, float* xout) {
  extern __shared__ char smem[];
  u16* Xs = (u16*)smem;                 // 64 x 192 bf16
  u16* Hs = Xs + 64 * CD;               // 64 x 768 bf16
  float* Y = (float*)(Hs + 64 * 4 * CD);  // 64 x 192 f32
  float* mu = Y + 64 * CD;
  float* rs = mu + 64;
  const int row0 = blockIdx.x * 64;
  const int tid = threadIdx.x, lane = tid & 31, wid = tid >> 5;
  const int nwv = blockDim.x >> 5;

  for (int idx = tid; idx < 64 * CD; idx += blockDim.x)
    Xs[idx] = f2bf(xin[(size_t)(row0 + idx / CD) * CD + idx % CD]);
  __syncthreads();

  // GEMM1: 4 x 48 tiles, K = 192; epilogue exact GELU
  for (int tile = wid; tile < 4 * 48; tile += nwv) {
    int tr = tile / 48, tc = tile % 48;
    v8f acc = {};
#pragma unroll
    for (int kk = 0; kk < 6; ++kk) {
      v16bf a = load_a_rm(Xs, tr * 16, kk * 32, CD, lane);
      v16bf b = load_b_nk(W1, tc * 16, kk * 32, CD, lane);
      acc = wmma_bf16(a, b, acc);
    }
#pragma unroll
    for (int r = 0; r < 8; ++r) {
      int n = tr * 16 + r + ((lane >> 4) << 3);
      int c = tc * 16 + (lane & 15);
      float v = acc[r] + B1[c];
      v = 0.5f * v * (1.0f + erff(v * 0.7071067811865475f));
      Hs[n * 4 * CD + c] = f2bf(v);
    }
  }
  __syncthreads();

  // GEMM2: 4 x 12 tiles, K = 768
  for (int tile = wid; tile < 4 * 12; tile += nwv) {
    int tr = tile / 12, tc = tile % 12;
    v8f acc = {};
#pragma unroll
    for (int kk = 0; kk < 24; ++kk) {
      v16bf a = load_a_rm(Hs, tr * 16, kk * 32, 4 * CD, lane);
      v16bf b = load_b_nk(W2, tc * 16, kk * 32, 4 * CD, lane);
      acc = wmma_bf16(a, b, acc);
    }
#pragma unroll
    for (int r = 0; r < 8; ++r) {
      int n = tr * 16 + r + ((lane >> 4) << 3);
      int c = tc * 16 + (lane & 15);
      Y[n * CD + c] = acc[r] + B2[c];
    }
  }
  __syncthreads();
  if (tid < 64) {
    float m = 0.0f;
    for (int c = 0; c < CD; ++c) m += Y[tid * CD + c];
    m *= (1.0f / CD);
    float v = 0.0f;
    for (int c = 0; c < CD; ++c) {
      float d = Y[tid * CD + c] - m;
      v += d * d;
    }
    mu[tid] = m;
    rs[tid] = rsqrtf(v * (1.0f / CD) + 1e-5f);
  }
  __syncthreads();
  for (int idx = tid; idx < 64 * CD; idx += blockDim.x) {
    int n = idx / CD, c = idx % CD;
    size_t gi = (size_t)(row0 + n) * CD + c;
    xout[gi] = xin[gi] + (Y[idx] - mu[n]) * rs[n] * g[c] + bb[c];
  }
}

// ---------------------------------------------------------------------------
extern "C" void kernel_launch(void* const* d_in, const int* in_sizes, int n_in,
                              void* d_out, int out_size, void* d_ws, size_t ws_size,
                              hipStream_t stream) {
  const float* x_in = (const float*)d_in[0];
  const float* qkv_w = (const float*)d_in[1];
  const float* qkv_b = (const float*)d_in[2];
  const float* proj_w = (const float*)d_in[3];
  const float* proj_b = (const float*)d_in[4];
  const float* bias_t = (const float*)d_in[5];
  const float* n1_g = (const float*)d_in[6];
  const float* n1_b = (const float*)d_in[7];
  const float* n2_g = (const float*)d_in[8];
  const float* n2_b = (const float*)d_in[9];
  const float* w1 = (const float*)d_in[10];
  const float* b1 = (const float*)d_in[11];
  const float* w2 = (const float*)d_in[12];
  const float* b2 = (const float*)d_in[13];

  float* xbuf = (float*)d_out;  // running activation buffer == output

  // workspace carve-up (~133 MB)
  char* ws = (char*)d_ws;
  size_t off = 0;
  u16* qs = (u16*)(ws + off);  off += (size_t)WINTOT * NHEAD * NWIN * HDIM * 2;
  u16* ks = (u16*)(ws + off);  off += (size_t)WINTOT * NHEAD * NWIN * HDIM * 2;
  u16* vtp = (u16*)(ws + off); off += (size_t)WINTOT * NHEAD * HDIM * NWP * 2;
  float* biasM = (float*)(ws + off); off += (size_t)TWIN * NHEAD * NWIN * NWIN * 4;
  u16* ao = (u16*)(ws + off);  off += (size_t)WINTOT * NWIN * CD * 2;
  u16* pkQ = (u16*)(ws + off); off += (size_t)3 * CD * CD * 2;      // [576][192]
  u16* pkP = (u16*)(ws + off); off += (size_t)CD * CD * 2;          // [192][192]
  u16* pk1 = (u16*)(ws + off); off += (size_t)4 * CD * CD * 2;      // [768][192]
  u16* pk2 = (u16*)(ws + off); off += (size_t)4 * CD * CD * 2;      // [192][768]
  (void)off; (void)ws_size; (void)n_in; (void)in_sizes; (void)out_size;

  k_copy<<<4096, 256, 0, stream>>>(xbuf, x_in, NTOK * CD);

  const size_t qkv_lds = (size_t)NWIN * CD * 2;               // 55.3 KB
  const size_t attn_lds = (size_t)9 * 16 * NWP * 2;           // 46.1 KB
  const size_t proj_lds = (size_t)NWIN * CD * 4 + NWIN * 8;   // 111.7 KB
  const size_t mlp_lds = (size_t)64 * CD * 2 + (size_t)64 * 4 * CD * 2 +
                         (size_t)64 * CD * 4 + 64 * 8;        // 172.5 KB

  for (int i = 0; i < 2; ++i) {
    const float* Wq = qkv_w + (size_t)i * CD * 3 * CD;
    const float* Bq = qkv_b + (size_t)i * 3 * CD;
    const float* Wp = proj_w + (size_t)i * CD * CD;
    const float* Bp = proj_b + (size_t)i * CD;
    const float* BT = bias_t + (size_t)i * 3312 * TWIN * NHEAD;
    const float* G1 = n1_g + (size_t)i * CD;
    const float* Bn1 = n1_b + (size_t)i * CD;
    const float* G2 = n2_g + (size_t)i * CD;
    const float* Bn2 = n2_b + (size_t)i * CD;
    const float* MW1 = w1 + (size_t)i * CD * 4 * CD;
    const float* MB1 = b1 + (size_t)i * 4 * CD;
    const float* MW2 = w2 + (size_t)i * 4 * CD * CD;
    const float* MB2 = b2 + (size_t)i * CD;
    const bool rolled = (i == 1);

    // pack weights into WMMA-B bf16 layout (once per layer)
    k_packw<<<432, 256, 0, stream>>>(Wq, pkQ, CD, 3 * CD);
    k_packw<<<144, 256, 0, stream>>>(Wp, pkP, CD, CD);
    k_packw<<<576, 256, 0, stream>>>(MW1, pk1, CD, 4 * CD);
    k_packw<<<576, 256, 0, stream>>>(MW2, pk2, 4 * CD, CD);

    if (!rolled)
      k_qkv<false><<<WINTOT, 256, qkv_lds, stream>>>(xbuf, pkQ, Bq, qs, ks, vtp);
    else
      k_qkv<true><<<WINTOT, 256, qkv_lds, stream>>>(xbuf, pkQ, Bq, qs, ks, vtp);

    k_bias<<<TWIN * NHEAD, 256, 0, stream>>>(BT, biasM, rolled ? 1 : 0);

    k_attn<<<WINTOT * NHEAD, 288, attn_lds, stream>>>(qs, ks, vtp, biasM, ao);

    if (!rolled)
      k_proj<false><<<WINTOT, 256, proj_lds, stream>>>(ao, pkP, Bp, G1, Bn1, xbuf);
    else
      k_proj<true><<<WINTOT, 256, proj_lds, stream>>>(ao, pkP, Bp, G1, Bn1, xbuf);

    k_mlp<<<NTOK / 64, 256, mlp_lds, stream>>>(xbuf, pk1, MB1, pk2, MB2, G2, Bn2,
                                               xbuf);
  }
}
